// _Joiner_34565896798556
// MI455X (gfx1250) — matrix operand order repaired and checked by reading, not compiled, not run on hardware
//
#include <hip/hip_runtime.h>
#include <hip/hip_bf16.h>
#include <stdint.h>

// Problem dims
#define BB 4
#define TT 256
#define UU 96
#define DD 512
#define VV 1024
#define M_PER_BATCH (TT*UU)          // 24576
#define M_TOTAL (BB*M_PER_BATCH)     // 98304

// Tiling
#define MBLK 256
#define NBLK 128
#define KC   32
#define NK   (DD/KC)                 // 16 k-steps
#define LDSP 40   // row pitch in bf16 elements (32 data + 8 pad), 80 B -> 16B aligned

#define W_ELEMS (VV*DD)              // 524288
#define WS_NEED (2u * W_ELEMS * 2u)  // hi+lo bf16 = 2 MB

typedef __attribute__((ext_vector_type(16))) __bf16    v16bf;
typedef __attribute__((ext_vector_type(8)))  float     v8f;
typedef __attribute__((ext_vector_type(4)))  float     f32x4;
typedef __attribute__((ext_vector_type(4)))  uint32_t  u32x4;
typedef __attribute__((ext_vector_type(8)))  uint32_t  u32x8;

// round-to-nearest-even f32 -> bf16 bits
static __device__ __forceinline__ unsigned short f2bf_rn(float f) {
    uint32_t x = __float_as_uint(f);
    uint32_t r = x + 0x7fffu + ((x >> 16) & 1u);
    return (unsigned short)(r >> 16);
}

// split x into hi (truncated top 16 bits, exact) and lo (bf16 of remainder)
static __device__ __forceinline__ void split_bf16(float x, unsigned short& hi, unsigned short& lo) {
    uint32_t xb = __float_as_uint(x);
    uint32_t hb = xb & 0xffff0000u;
    hi = (unsigned short)(xb >> 16);
    lo = f2bf_rn(x - __uint_as_float(hb));
}

// build a v16bf fragment from two 16-byte LDS runs (run2 at +off2 ushorts)
static __device__ __forceinline__ v16bf make_frag(const unsigned short* p, int off2) {
    u32x4 r1 = *(const u32x4*)(p);
    u32x4 r2 = *(const u32x4*)(p + off2);
    u32x8 c;
    c[0] = r1[0]; c[1] = r1[1]; c[2] = r1[2]; c[3] = r1[3];
    c[4] = r2[0]; c[5] = r2[1]; c[6] = r2[2]; c[7] = r2[3];
    return __builtin_bit_cast(v16bf, c);
}

// ---------------- W split prep: f32 (V,D) -> bf16 hi/lo planes in ws ----------------
__global__ __launch_bounds__(256)
void wsplit_kernel(const float* __restrict__ W,
                   unsigned short* __restrict__ wh, unsigned short* __restrict__ wl)
{
    int idx = (blockIdx.x * 256 + threadIdx.x) * 4;   // covers W_ELEMS with grid 512
    f32x4 w4 = *(const f32x4*)(W + idx);
    unsigned short h[4], l[4];
#pragma unroll
    for (int c = 0; c < 4; ++c) split_bf16(w4[c], h[c], l[c]);
    uint2 hp = make_uint2(((uint32_t)h[1] << 16) | h[0], ((uint32_t)h[3] << 16) | h[2]);
    uint2 lp = make_uint2(((uint32_t)l[1] << 16) | l[0], ((uint32_t)l[3] << 16) | l[2]);
    *(uint2*)(wh + idx) = hp;
    *(uint2*)(wl + idx) = lp;
}

// ---------------- main GEMM ----------------
// USE_WS: W already split to bf16 hi/lo planes (wh/wl); else split inline from f32 W.
template <bool USE_WS>
__global__ __launch_bounds__(256, 1)
void joiner_gemm(const float* __restrict__ src, const float* __restrict__ tgt,
                 const float* __restrict__ W,   const float* __restrict__ bias,
                 const unsigned short* __restrict__ wh, const unsigned short* __restrict__ wl,
                 float* __restrict__ out)
{
    // double-buffered tiles
    __shared__ unsigned short Ahi[2][MBLK * LDSP];
    __shared__ unsigned short Alo[2][MBLK * LDSP];
    __shared__ unsigned short Whi[2][NBLK * LDSP];
    __shared__ unsigned short Wlo[2][NBLK * LDSP];

    const int tid   = threadIdx.x;
    const int lane  = tid & 31;
    const int wave  = tid >> 5;
    const int waveM = wave & 3;    // 0..3 -> 64-row strip
    const int waveN = wave >> 2;   // 0..1 -> 64-col strip

    const int v0  = blockIdx.x * NBLK;
    const int m0  = blockIdx.y * MBLK;
    const int b   = m0 / M_PER_BATCH;     // MBLK divides M_PER_BATCH -> no straddle
    const int m0b = m0 % M_PER_BATCH;

    const float* __restrict__ srcB = src + (size_t)b * TT * DD;
    const float* __restrict__ tgtB = tgt + (size_t)b * UU * DD;

    const int fr = lane & 15;
    const int fh = lane >> 4;

    // ---- staging lambda: fill buffer p with k-chunk kk
    auto stage = [&](int kk, int p) {
        const int kbase = kk * KC;
        // A tile: relu(src+tgt), split hi/lo (256 rows x 32 k)
#pragma unroll
        for (int i = 0; i < 8; ++i) {
            int tsk = tid + i * 256;
            int r   = tsk >> 3;
            int kg  = (tsk & 7) << 2;
            int mrow = m0b + r;
            int t = mrow / UU;
            int u = mrow - t * UU;
            f32x4 s4 = *(const f32x4*)(srcB + t * DD + kbase + kg);
            f32x4 g4 = *(const f32x4*)(tgtB + u * DD + kbase + kg);
            unsigned short h[4], l[4];
#pragma unroll
            for (int c = 0; c < 4; ++c) {
                float v = s4[c] + g4[c];
                v = v > 0.0f ? v : 0.0f;
                split_bf16(v, h[c], l[c]);
            }
            uint2 hp = make_uint2(((uint32_t)h[1] << 16) | h[0], ((uint32_t)h[3] << 16) | h[2]);
            uint2 lp = make_uint2(((uint32_t)l[1] << 16) | l[0], ((uint32_t)l[3] << 16) | l[2]);
            *(uint2*)(&Ahi[p][r * LDSP + kg]) = hp;
            *(uint2*)(&Alo[p][r * LDSP + kg]) = lp;
        }
        // W tile (128 cols x 32 k)
        if (USE_WS) {
            // pure copy of pre-split bf16 planes: 512 16B-chunk tasks per plane
#pragma unroll
            for (int i = 0; i < 2; ++i) {
                int tsk = tid + i * 256;            // 0..511
                int r   = tsk >> 2;                 // 0..127
                int kg8 = (tsk & 3) << 3;           // 0,8,16,24
                size_t gofs = (size_t)(v0 + r) * DD + kbase + kg8;
                *(u32x4*)(&Whi[p][r * LDSP + kg8]) = *(const u32x4*)(wh + gofs);
                *(u32x4*)(&Wlo[p][r * LDSP + kg8]) = *(const u32x4*)(wl + gofs);
            }
        } else {
#pragma unroll
            for (int i = 0; i < 4; ++i) {
                int tsk = tid + i * 256;
                int r   = tsk >> 3;
                int kg  = (tsk & 7) << 2;
                f32x4 w4 = *(const f32x4*)(W + (size_t)(v0 + r) * DD + kbase + kg);
                unsigned short h[4], l[4];
#pragma unroll
                for (int c = 0; c < 4; ++c) split_bf16(w4[c], h[c], l[c]);
                uint2 hp = make_uint2(((uint32_t)h[1] << 16) | h[0], ((uint32_t)h[3] << 16) | h[2]);
                uint2 lp = make_uint2(((uint32_t)l[1] << 16) | l[0], ((uint32_t)l[3] << 16) | l[2]);
                *(uint2*)(&Whi[p][r * LDSP + kg]) = hp;
                *(uint2*)(&Wlo[p][r * LDSP + kg]) = lp;
            }
        }
    };

    v8f acc[4][4];
#pragma unroll
    for (int i = 0; i < 4; ++i)
#pragma unroll
        for (int j = 0; j < 4; ++j)
            acc[i][j] = (v8f)(0.0f);

    float biasv[4];
#pragma unroll
    for (int j = 0; j < 4; ++j)
        biasv[j] = bias[v0 + waveN * 64 + j * 16 + fr];

    // prologue: stage chunk 0
    stage(0, 0);
    __syncthreads();

    for (int kk = 0; kk < NK; ++kk) {
        const int cur = kk & 1;

        // ---- fragments + WMMAs from LDS[cur]
        v16bf aHi[4], aLo[4];
#pragma unroll
        for (int i = 0; i < 4; ++i) {
            int row = waveM * 64 + i * 16 + fr;
            aHi[i] = make_frag(&Ahi[cur][row * LDSP + fh * 8], 16);
            aLo[i] = make_frag(&Alo[cur][row * LDSP + fh * 8], 16);
        }
#pragma unroll
        for (int j = 0; j < 4; ++j) {
            int col = waveN * 64 + j * 16 + fr;
            v16bf bHi = make_frag(&Whi[cur][col * LDSP + fh * 8], 16);
            v16bf bLo = make_frag(&Wlo[cur][col * LDSP + fh * 8], 16);
#pragma unroll
            for (int i = 0; i < 4; ++i) {
                acc[i][j] = __builtin_amdgcn_wmma_f32_16x16x32_bf16(
                    false, aHi[i], false, bHi, (short)0, acc[i][j], false, false);
                acc[i][j] = __builtin_amdgcn_wmma_f32_16x16x32_bf16(
                    false, aHi[i], false, bLo, (short)0, acc[i][j], false, false);
                acc[i][j] = __builtin_amdgcn_wmma_f32_16x16x32_bf16(
                    false, aLo[i], false, bHi, (short)0, acc[i][j], false, false);
            }
        }

        // ---- stage next chunk into the other buffer (overlaps XDL work above)
        if (kk + 1 < NK)
            stage(kk + 1, cur ^ 1);

        __syncthreads();
    }

    // ---- epilogue: bias add + store (D layout: lane -> n=fr, m = 8*fh + r)
#pragma unroll
    for (int i = 0; i < 4; ++i) {
        int mg = m0 + waveM * 64 + i * 16 + fh * 8;
#pragma unroll
        for (int j = 0; j < 4; ++j) {
            int vg = v0 + waveN * 64 + j * 16 + fr;
            float bv = biasv[j];
            float* p = out + (size_t)mg * VV + vg;
#pragma unroll
            for (int r = 0; r < 8; ++r)
                p[(size_t)r * VV] = acc[i][j][r] + bv;
        }
    }
}

__global__ void copy_lengths(const int* __restrict__ sl, const int* __restrict__ tl,
                             int* __restrict__ dst)
{
    int i = threadIdx.x;
    if (i < 4)      dst[i] = sl[i];
    else if (i < 8) dst[i] = tl[i - 4];
}

extern "C" void kernel_launch(void* const* d_in, const int* in_sizes, int n_in,
                              void* d_out, int out_size, void* d_ws, size_t ws_size,
                              hipStream_t stream) {
    const float* src  = (const float*)d_in[0];
    const int*   slen = (const int*)  d_in[1];
    const float* tgt  = (const float*)d_in[2];
    const int*   tlen = (const int*)  d_in[3];
    const float* W    = (const float*)d_in[4];
    const float* bias = (const float*)d_in[5];
    float* out = (float*)d_out;

    dim3 grid(VV / NBLK, M_TOTAL / MBLK);   // (8, 384)

    if (ws_size >= WS_NEED && d_ws != nullptr) {
        unsigned short* wh = (unsigned short*)d_ws;
        unsigned short* wl = wh + W_ELEMS;
        wsplit_kernel<<<W_ELEMS / (256 * 4), 256, 0, stream>>>(W, wh, wl);
        joiner_gemm<true><<<grid, 256, 0, stream>>>(src, tgt, W, bias, wh, wl, out);
    } else {
        joiner_gemm<false><<<grid, 256, 0, stream>>>(src, tgt, W, bias, nullptr, nullptr, out);
    }

    copy_lengths<<<1, 8, 0, stream>>>(slen, tlen, (int*)out + (size_t)M_TOTAL * VV);
}
